// GAT_1090921693772
// MI455X (gfx1250) — compile-verified, matrix operand order checked by reading, not run on hardware
//
#include <hip/hip_runtime.h>
#include <hip/hip_bf16.h>

typedef __attribute__((ext_vector_type(16))) _Float16 v16h;
typedef __attribute__((ext_vector_type(8)))  _Float16 v8h;
typedef __attribute__((ext_vector_type(8)))  float    v8f;

#define IN_DIM   768
#define HID      128
#define HEADS    8
#define CPH      16      // channels per head
#define NCOMB    256     // combined [W1 | Wp] output columns
#define OUT_DIM  2
#define TILE_K   32
#define KITERS   (IN_DIM / TILE_K)
#define NEG_SLOPE 0.2f
#define EPS      1e-16f

// ---------------------------------------------------------------- utilities

__device__ __forceinline__ void atomicMaxFloat(float* addr, float v) {
    // classic monotone trick; accumulator initialized to -inf (0xFF800000)
    if (v >= 0.0f) atomicMax((int*)addr, __float_as_int(v));
    else           atomicMin((unsigned int*)addr, __float_as_uint(v));
}

__device__ __forceinline__ void edge_sd(const int* __restrict__ ei, int E, int e,
                                        int& s, int& d) {
    if (e < E) { s = ei[e]; d = ei[E + e]; }
    else       { s = d = e - E; }          // appended self loops
}

// Low 32 bits of a flat shared-aperture pointer == LDS byte offset.
__device__ __forceinline__ unsigned lds_off(const void* p) {
    return (unsigned)(unsigned long long)p;
}

// gfx1250 async global->LDS copy, 16B per lane, tracked by ASYNCcnt.
__device__ __forceinline__ void async_b128(unsigned lds, const void* g) {
    asm volatile("global_load_async_to_lds_b128 %0, %1, off"
                 :: "v"(lds), "v"((unsigned long long)g) : "memory");
}
__device__ __forceinline__ void wait_async0() {
    asm volatile("s_wait_asynccnt 0x0" ::: "memory");
}

// ---------------------------------------------------------------- weights pack
// WcT[n][k] = (n<128 ? W1[k][n] : Wp[k][n-128]) as f16
__global__ void pack_w_kernel(const float* __restrict__ W1,
                              const float* __restrict__ Wp,
                              _Float16* __restrict__ wT) {
    int i = blockIdx.x * blockDim.x + threadIdx.x;
    if (i >= NCOMB * IN_DIM) return;
    int n = i / IN_DIM, k = i - n * IN_DIM;
    float v = (n < HID) ? W1[(size_t)k * HID + n] : Wp[(size_t)k * HID + (n - HID)];
    wT[i] = (_Float16)v;
}

// ---------------------------------------------------------------- fused GEMM
// Hc = x[N x 768] * Wc[768 x 256]; cols 0..127 -> h1, 128..255 -> hp.
// 256 threads = 8 waves; block tile 128(M) x 128(N); wave tile 32x64 (2x4 WMMA).
// Double-buffered K loop fed entirely by async global->LDS copies.
__global__ __launch_bounds__(256) void gemm1_wmma_kernel(
        const float* __restrict__ x, const _Float16* __restrict__ wT,
        float* __restrict__ h1, float* __restrict__ hp) {
    __shared__ float    As[2][128 * TILE_K];   // A kept f32 in LDS (async, no staging)
    __shared__ _Float16 Bs[2][128 * TILE_K];

    const int tid  = threadIdx.x;
    const int wave = tid >> 5;
    const int lane = tid & 31;
    const int wm   = (wave & 3) * 32;          // wave M offset in block tile
    const int wn   = (wave >> 2) * 64;         // wave N offset in block tile
    const int m0   = blockIdx.y * 128;
    const int n0   = blockIdx.x * 128;

    const int lr = lane & 15;                  // row (A) / col (B) within tile
    const int kg = lane >> 4;                  // K-group select (ISA A/B layout)

    v8f acc[2][4] = {};

    // issue this thread's async slice of one K tile into buffer `buf`
    auto issue_tile = [&](int k0, int buf) {
        // A: 128 rows x 32 f32 = 1024 x 16B segments, 4 per thread
        #pragma unroll
        for (int q = 0; q < 4; ++q) {
            int s   = tid + q * 256;
            int row = s >> 3, c4 = s & 7;      // 16B = 4 floats
            async_b128(lds_off(&As[buf][row * TILE_K + c4 * 4]),
                       x + (size_t)(m0 + row) * IN_DIM + k0 + c4 * 4);
        }
        // B: 128 cols x 32 f16 = 512 x 16B segments, 2 per thread
        #pragma unroll
        for (int q = 0; q < 2; ++q) {
            int s = tid + q * 256;
            int n = s >> 2, c8 = s & 3;        // 16B = 8 halfs
            async_b128(lds_off(&Bs[buf][n * TILE_K + c8 * 8]),
                       wT + (size_t)(n0 + n) * IN_DIM + k0 + c8 * 8);
        }
    };

    issue_tile(0, 0);

    for (int it = 0; it < KITERS; ++it) {
        wait_async0();        // my async writes done
        __syncthreads();      // everyone's async writes visible
        if (it + 1 < KITERS) issue_tile((it + 1) * TILE_K, (it + 1) & 1);

        const float*    Ab = As[it & 1];
        const _Float16* Bb = Bs[it & 1];

        // ---- A fragments: f32 LDS -> f16, ISA layout (K chunks kg*8, 16+kg*8)
        v16h afrag[2];
        #pragma unroll
        for (int i = 0; i < 2; ++i) {
            const float* ap = Ab + (wm + i * 16 + lr) * TILE_K + kg * 8;
            float4 u0 = *(const float4*)(ap + 0);
            float4 u1 = *(const float4*)(ap + 4);
            float4 u2 = *(const float4*)(ap + 16);
            float4 u3 = *(const float4*)(ap + 20);
            v16h a;
            a[0]  = (_Float16)u0.x; a[1]  = (_Float16)u0.y;
            a[2]  = (_Float16)u0.z; a[3]  = (_Float16)u0.w;
            a[4]  = (_Float16)u1.x; a[5]  = (_Float16)u1.y;
            a[6]  = (_Float16)u1.z; a[7]  = (_Float16)u1.w;
            a[8]  = (_Float16)u2.x; a[9]  = (_Float16)u2.y;
            a[10] = (_Float16)u2.z; a[11] = (_Float16)u2.w;
            a[12] = (_Float16)u3.x; a[13] = (_Float16)u3.y;
            a[14] = (_Float16)u3.z; a[15] = (_Float16)u3.w;
            afrag[i] = a;
        }
        // ---- B fragments
        v16h bfrag[4];
        #pragma unroll
        for (int j = 0; j < 4; ++j) {
            const _Float16* bp = Bb + (wn + j * 16 + lr) * TILE_K + kg * 8;
            v8h lo = *(const v8h*)bp;
            v8h hi = *(const v8h*)(bp + 16);
            v16h b;
            #pragma unroll
            for (int t = 0; t < 8; ++t) { b[t] = lo[t]; b[8 + t] = hi[t]; }
            bfrag[j] = b;
        }
        #pragma unroll
        for (int i = 0; i < 2; ++i)
            #pragma unroll
            for (int j = 0; j < 4; ++j)
                acc[i][j] = __builtin_amdgcn_wmma_f32_16x16x32_f16(
                    false, afrag[i], false, bfrag[j],
                    (short)0, acc[i][j], false, false);
    }

    // ---- store: C layout: lane col = lane&15; VGPR r -> row kg*8 + r
    #pragma unroll
    for (int i = 0; i < 2; ++i)
        #pragma unroll
        for (int j = 0; j < 4; ++j) {
            int col = n0 + wn + j * 16 + lr;
            float* outp = (col < HID) ? h1 : hp;
            int   ccol  = (col < HID) ? col : col - HID;
            #pragma unroll
            for (int r = 0; r < 8; ++r) {
                int row = m0 + wm + i * 16 + kg * 8 + r;
                outp[(size_t)row * HID + ccol] = acc[i][j][r];
            }
        }
}

// ---------------------------------------------------------------- layer-1 attn scores
__global__ void attn1_kernel(const float* __restrict__ h1,
                             const float* __restrict__ att_s,
                             const float* __restrict__ att_d,
                             float* __restrict__ a_s, float* __restrict__ a_d, int N) {
    int i = blockIdx.x * blockDim.x + threadIdx.x;   // n*8 + h
    if (i >= N * HEADS) return;
    int h = i & (HEADS - 1);
    const float* hv = h1 + (size_t)(i >> 3) * HID + h * CPH;
    float ss = 0.f, sd = 0.f;
    #pragma unroll
    for (int c = 0; c < CPH; ++c) {
        float v = hv[c];
        ss += v * att_s[h * CPH + c];
        sd += v * att_d[h * CPH + c];
    }
    a_s[i] = ss; a_d[i] = sd;
}

__global__ void init1_kernel(float* __restrict__ m, float* __restrict__ den,
                             float* __restrict__ out1, int N) {
    int i = blockIdx.x * blockDim.x + threadIdx.x;
    if (i < N * HEADS) { m[i] = -__builtin_inff(); den[i] = 0.f; }
    if (i < N * HID)   out1[i] = 0.f;
}

// ---------------------------------------------------------------- layer-1 edge passes
__global__ void edge1_max_kernel(const int* __restrict__ ei, int E, int N,
                                 const float* __restrict__ a_s, const float* __restrict__ a_d,
                                 float* __restrict__ m) {
    int idx = blockIdx.x * blockDim.x + threadIdx.x;
    if (idx >= (E + N) * HEADS) return;
    int e = idx >> 3, h = idx & 7, s, d;
    edge_sd(ei, E, e, s, d);
    float v = a_s[s * HEADS + h] + a_d[d * HEADS + h];
    v = (v > 0.f) ? v : NEG_SLOPE * v;
    atomicMaxFloat(&m[d * HEADS + h], v);
}

__global__ void edge1_sum_kernel(const int* __restrict__ ei, int E, int N,
                                 const float* __restrict__ a_s, const float* __restrict__ a_d,
                                 const float* __restrict__ m, float* __restrict__ den) {
    int idx = blockIdx.x * blockDim.x + threadIdx.x;
    if (idx >= (E + N) * HEADS) return;
    int e = idx >> 3, h = idx & 7, s, d;
    edge_sd(ei, E, e, s, d);
    float v = a_s[s * HEADS + h] + a_d[d * HEADS + h];
    v = (v > 0.f) ? v : NEG_SLOPE * v;
    atomicAdd(&den[d * HEADS + h], __expf(v - m[d * HEADS + h]));
}

__global__ void edge1_agg_kernel(const int* __restrict__ ei, int E, int N,
                                 const float* __restrict__ a_s, const float* __restrict__ a_d,
                                 const float* __restrict__ m, const float* __restrict__ den,
                                 const float* __restrict__ h1, float* __restrict__ out1) {
    int idx = blockIdx.x * blockDim.x + threadIdx.x;
    if (idx >= (E + N) * HEADS) return;
    int e = idx >> 3, h = idx & 7, s, d;
    edge_sd(ei, E, e, s, d);
    float v = a_s[s * HEADS + h] + a_d[d * HEADS + h];
    v = (v > 0.f) ? v : NEG_SLOPE * v;
    float alpha = __expf(v - m[d * HEADS + h]) / (den[d * HEADS + h] + EPS);
    const float* hs = h1   + (size_t)s * HID + h * CPH;
    float*       od = out1 + (size_t)d * HID + h * CPH;
    #pragma unroll
    for (int c = 0; c < CPH; ++c)
        atomicAdd(&od[c], hs[c] * alpha);
}

// ---------------------------------------------------------------- mid node update
// h = elu(out1 + b1 + hp + bp); h2 = h @ W2; layer-2 scores; init layer-2 accums
__global__ void node_mid_kernel(const float* __restrict__ out1, const float* __restrict__ hp,
                                const float* __restrict__ b1,   const float* __restrict__ bp,
                                const float* __restrict__ W2,
                                const float* __restrict__ aw_s, const float* __restrict__ aw_d,
                                float* __restrict__ h2, float* __restrict__ as2,
                                float* __restrict__ ad2, float* __restrict__ m2,
                                float* __restrict__ den2, float* __restrict__ out2, int N) {
    int n = blockIdx.x * blockDim.x + threadIdx.x;
    if (n >= N) return;
    float acc0 = 0.f, acc1 = 0.f;
    const float* o = out1 + (size_t)n * HID;
    const float* p = hp   + (size_t)n * HID;
    for (int k = 0; k < HID; ++k) {
        float v = o[k] + b1[k] + p[k] + bp[k];
        v = (v > 0.f) ? v : (__expf(v) - 1.f);           // ELU(alpha=1)
        acc0 += v * W2[k * OUT_DIM + 0];
        acc1 += v * W2[k * OUT_DIM + 1];
    }
    h2[n * OUT_DIM + 0] = acc0;
    h2[n * OUT_DIM + 1] = acc1;
    as2[n] = acc0 * aw_s[0] + acc1 * aw_s[1];
    ad2[n] = acc0 * aw_d[0] + acc1 * aw_d[1];
    m2[n] = -__builtin_inff(); den2[n] = 0.f;
    out2[n * OUT_DIM + 0] = 0.f; out2[n * OUT_DIM + 1] = 0.f;
}

// ---------------------------------------------------------------- layer-2 edge passes
__global__ void edge2_max_kernel(const int* __restrict__ ei, int E, int N,
                                 const float* __restrict__ as2, const float* __restrict__ ad2,
                                 float* __restrict__ m2) {
    int e = blockIdx.x * blockDim.x + threadIdx.x;
    if (e >= E + N) return;
    int s, d; edge_sd(ei, E, e, s, d);
    float v = as2[s] + ad2[d];
    v = (v > 0.f) ? v : NEG_SLOPE * v;
    atomicMaxFloat(&m2[d], v);
}

__global__ void edge2_sum_kernel(const int* __restrict__ ei, int E, int N,
                                 const float* __restrict__ as2, const float* __restrict__ ad2,
                                 const float* __restrict__ m2, float* __restrict__ den2) {
    int e = blockIdx.x * blockDim.x + threadIdx.x;
    if (e >= E + N) return;
    int s, d; edge_sd(ei, E, e, s, d);
    float v = as2[s] + ad2[d];
    v = (v > 0.f) ? v : NEG_SLOPE * v;
    atomicAdd(&den2[d], __expf(v - m2[d]));
}

__global__ void edge2_agg_kernel(const int* __restrict__ ei, int E, int N,
                                 const float* __restrict__ as2, const float* __restrict__ ad2,
                                 const float* __restrict__ m2, const float* __restrict__ den2,
                                 const float* __restrict__ h2, float* __restrict__ out2) {
    int e = blockIdx.x * blockDim.x + threadIdx.x;
    if (e >= E + N) return;
    int s, d; edge_sd(ei, E, e, s, d);
    float v = as2[s] + ad2[d];
    v = (v > 0.f) ? v : NEG_SLOPE * v;
    float alpha = __expf(v - m2[d]) / (den2[d] + EPS);
    atomicAdd(&out2[d * OUT_DIM + 0], h2[s * OUT_DIM + 0] * alpha);
    atomicAdd(&out2[d * OUT_DIM + 1], h2[s * OUT_DIM + 1] * alpha);
}

__global__ void final_kernel(const float* __restrict__ out2, const float* __restrict__ b2,
                             float* __restrict__ y, int N) {
    int i = blockIdx.x * blockDim.x + threadIdx.x;
    if (i >= N * OUT_DIM) return;
    y[i] = out2[i] + b2[i & 1];          // mean over 1 head == identity
}

// ---------------------------------------------------------------- host launcher
extern "C" void kernel_launch(void* const* d_in, const int* in_sizes, int n_in,
                              void* d_out, int out_size, void* d_ws, size_t ws_size,
                              hipStream_t stream) {
    const float* x     = (const float*)d_in[0];
    const int*   ei    = (const int*)  d_in[1];
    const float* W1    = (const float*)d_in[2];
    const float* atS1  = (const float*)d_in[3];
    const float* atD1  = (const float*)d_in[4];
    const float* b1    = (const float*)d_in[5];
    const float* Wp    = (const float*)d_in[6];
    const float* bp    = (const float*)d_in[7];
    const float* W2    = (const float*)d_in[8];
    const float* atS2  = (const float*)d_in[9];
    const float* atD2  = (const float*)d_in[10];
    const float* b2    = (const float*)d_in[11];
    float* y = (float*)d_out;

    const int N = in_sizes[0] / IN_DIM;
    const int E = in_sizes[1] / 2;

    // ---- workspace carve-out (all 256B aligned)
    char* ws = (char*)d_ws;
    size_t off = 0;
    auto carve = [&](size_t bytes) { void* p = ws + off; off += (bytes + 255) & ~(size_t)255; return p; };
    _Float16* wT  = (_Float16*)carve((size_t)NCOMB * IN_DIM * 2);
    float* h1   = (float*)carve((size_t)N * HID * 4);
    float* hp   = (float*)carve((size_t)N * HID * 4);
    float* out1 = (float*)carve((size_t)N * HID * 4);
    float* a_s  = (float*)carve((size_t)N * HEADS * 4);
    float* a_d  = (float*)carve((size_t)N * HEADS * 4);
    float* mmax = (float*)carve((size_t)N * HEADS * 4);
    float* den  = (float*)carve((size_t)N * HEADS * 4);
    float* h2   = (float*)carve((size_t)N * OUT_DIM * 4);
    float* as2  = (float*)carve((size_t)N * 4);
    float* ad2  = (float*)carve((size_t)N * 4);
    float* m2   = (float*)carve((size_t)N * 4);
    float* den2 = (float*)carve((size_t)N * 4);
    float* out2 = (float*)carve((size_t)N * OUT_DIM * 4);

    const int T = 256;
    // 1) pack weights
    pack_w_kernel<<<(NCOMB * IN_DIM + T - 1) / T, T, 0, stream>>>(W1, Wp, wT);
    // 2) fused WMMA GEMM  Hc = x @ [W1|Wp]  (async double-buffered)
    dim3 gg(NCOMB / 128, N / 128);
    gemm1_wmma_kernel<<<gg, T, 0, stream>>>(x, wT, h1, hp);
    // 3) per-(node,head) scores
    attn1_kernel<<<(N * HEADS + T - 1) / T, T, 0, stream>>>(h1, atS1, atD1, a_s, a_d, N);
    // 4) init accumulators
    init1_kernel<<<(N * HID + T - 1) / T, T, 0, stream>>>(mmax, den, out1, N);
    // 5-7) layer-1 edge passes (segment softmax + weighted scatter)
    int EH = (E + N) * HEADS;
    edge1_max_kernel<<<(EH + T - 1) / T, T, 0, stream>>>(ei, E, N, a_s, a_d, mmax);
    edge1_sum_kernel<<<(EH + T - 1) / T, T, 0, stream>>>(ei, E, N, a_s, a_d, mmax, den);
    edge1_agg_kernel<<<(EH + T - 1) / T, T, 0, stream>>>(ei, E, N, a_s, a_d, mmax, den, h1, out1);
    // 8) node update: ELU(residual) -> h2, layer-2 scores, init layer-2 accums
    node_mid_kernel<<<(N + T - 1) / T, T, 0, stream>>>(out1, hp, b1, bp, W2, atS2, atD2,
                                                       h2, as2, ad2, m2, den2, out2, N);
    // 9-11) layer-2 edge passes
    int ET = E + N;
    edge2_max_kernel<<<(ET + T - 1) / T, T, 0, stream>>>(ei, E, N, as2, ad2, m2);
    edge2_sum_kernel<<<(ET + T - 1) / T, T, 0, stream>>>(ei, E, N, as2, ad2, m2, den2);
    edge2_agg_kernel<<<(ET + T - 1) / T, T, 0, stream>>>(ei, E, N, as2, ad2, m2, den2, h2, out2);
    // 12) bias -> output
    final_kernel<<<(N * OUT_DIM + T - 1) / T, T, 0, stream>>>(out2, b2, y, N);
}
